// PCQMContactGNNNode_22385369546873
// MI455X (gfx1250) — matrix operand order repaired and checked by reading, not compiled, not run on hardware
//
#include <hip/hip_runtime.h>
#include <hip/hip_bf16.h>

#define DIM      256
#define NLAYERS  6
#define BN_EPS   1e-5f
#define LDB      264   // padded LDS row stride in bf16 elements (528 B = 33*16 B)

typedef __bf16 v16bf __attribute__((ext_vector_type(16)));
typedef __bf16 v2bf  __attribute__((ext_vector_type(2)));
typedef float  v8f   __attribute__((ext_vector_type(8)));
typedef unsigned int u32x4 __attribute__((ext_vector_type(4)));
typedef int          i32x4 __attribute__((ext_vector_type(4)));
typedef int          i32x8 __attribute__((ext_vector_type(8)));

__device__ __forceinline__ __bf16 f2bf(float f) {
  unsigned u = __builtin_bit_cast(unsigned, f);
  u += 0x7FFFu + ((u >> 16) & 1u);            // round-to-nearest-even
  return __builtin_bit_cast(__bf16, (unsigned short)(u >> 16));
}

// Pack two fp32 into two bf16 (round-to-nearest) with a single v_perm_b32:
// result = { hi16(y+0x8000), hi16(x+0x8000) }.
__device__ __forceinline__ unsigned pkbf(float x, float y) {
  unsigned ux = __builtin_bit_cast(unsigned, x) + 0x8000u;
  unsigned uy = __builtin_bit_cast(unsigned, y) + 0x8000u;
  // sel 0x07060302: b0=S1.b2, b1=S1.b3 (x hi16), b2=S0.b2, b3=S0.b3 (y hi16)
  return __builtin_amdgcn_perm(uy, ux, 0x07060302u);
}

// Pack 16 fp32 -> v16bf A-fragment.
__device__ __forceinline__ v16bf cvt16(float4 f0, float4 f1, float4 f2, float4 f3) {
#if __has_builtin(__builtin_amdgcn_cvt_pk_bf16_f32)
  union { v16bf v; v2bf p[8]; } u;
  u.p[0] = __builtin_amdgcn_cvt_pk_bf16_f32(f0.x, f0.y);
  u.p[1] = __builtin_amdgcn_cvt_pk_bf16_f32(f0.z, f0.w);
  u.p[2] = __builtin_amdgcn_cvt_pk_bf16_f32(f1.x, f1.y);
  u.p[3] = __builtin_amdgcn_cvt_pk_bf16_f32(f1.z, f1.w);
  u.p[4] = __builtin_amdgcn_cvt_pk_bf16_f32(f2.x, f2.y);
  u.p[5] = __builtin_amdgcn_cvt_pk_bf16_f32(f2.z, f2.w);
  u.p[6] = __builtin_amdgcn_cvt_pk_bf16_f32(f3.x, f3.y);
  u.p[7] = __builtin_amdgcn_cvt_pk_bf16_f32(f3.z, f3.w);
  return u.v;
#else
  union { v16bf v; unsigned u[8]; } r;
  r.u[0] = pkbf(f0.x, f0.y);
  r.u[1] = pkbf(f0.z, f0.w);
  r.u[2] = pkbf(f1.x, f1.y);
  r.u[3] = pkbf(f1.z, f1.w);
  r.u[4] = pkbf(f2.x, f2.y);
  r.u[5] = pkbf(f2.z, f2.w);
  r.u[6] = pkbf(f3.x, f3.y);
  r.u[7] = pkbf(f3.z, f3.w);
  return r.v;
#endif
}

// TDM: DMA the packed 256x256 bf16 weight matrix (K-contiguous rows) into LDS,
// inserting 16 B of padding per 512 B row -> LDS row stride 264 elements.
__device__ __forceinline__ void tdm_load_weights(const __bf16* Bt, unsigned lds_off) {
  unsigned long long ga = (unsigned long long)(uintptr_t)Bt;
  u32x4 g0;
  g0[0] = 1u;                                                  // count=1 (user D#)
  g0[1] = lds_off;                                             // lds_addr
  g0[2] = (unsigned)ga;                                        // global_addr[31:0]
  g0[3] = (unsigned)((ga >> 32) & 0x01FFFFFFu) | (2u << 30);   // addr[56:32], type=2
  i32x8 g1;
  g1[0] = (1 << 16) | (1 << 20) | (6 << 22) | (3 << 25);       // 2B elems, pad_en,
                                                               // interval=512B, amount=4dw
  g1[1] = (int)(256u << 16);   // tensor_dim0 = 256 (low 16 in [31:16])
  g1[2] = (int)(256u << 16);   // tensor_dim0 hi = 0 ; tensor_dim1 = 256 (low 16)
  g1[3] = (int)(256u << 16);   // tensor_dim1 hi = 0 ; tile_dim0 = 256
  g1[4] = 256;                 // tile_dim1 = 256 ; tile_dim2 = 0
  g1[5] = 256;                 // tensor_dim0_stride low32 = 256
  g1[6] = 0;
  g1[7] = 0;
  i32x4 gz = {0, 0, 0, 0};
#if defined(__clang_major__) && __clang_major__ >= 23
  i32x8 gz8 = {0, 0, 0, 0, 0, 0, 0, 0};
  __builtin_amdgcn_tensor_load_to_lds(g0, g1, gz, gz, gz8, 0);
#else
  __builtin_amdgcn_tensor_load_to_lds(g0, g1, gz, gz, 0);
#endif
  __builtin_amdgcn_s_wait_tensorcnt(0);
}

// ---------------------------------------------------------------- utilities
__global__ void copy_f4(const float4* __restrict__ src, float4* __restrict__ dst, int n4) {
  int i = blockIdx.x * blockDim.x + threadIdx.x;
  if (i < n4) dst[i] = src[i];
}

__global__ void zero_stats(float* __restrict__ s) {   // 1 block, 1024 threads
  s[threadIdx.x] = 0.0f;
}

// fp32 W[l][k][n] (row-major) -> bf16 Wt[l][n][k] (K-contiguous).
__global__ void pack_weights(const float* __restrict__ W, __bf16* __restrict__ Wt) {
  int id = blockIdx.x * blockDim.x + threadIdx.x;     // = l*65536 + k*256 + n
  int l = id >> 16, rem = id & 0xFFFF, k = rem >> 8, n = rem & 0xFF;
  Wt[(size_t)l * DIM * DIM + (size_t)n * DIM + k] = f2bf(W[id]);
}

// ---------------------------------------------------------------- scatter-add
__global__ void scatter_add(const float* __restrict__ h, const long long* __restrict__ ei,
                            float* __restrict__ z, int E) {
  int gid  = blockIdx.x * blockDim.x + threadIdx.x;
  int e    = gid >> 5;
  int lane = gid & 31;
  if (e >= E) return;
  long long s = ei[e];                 // src
  long long d = ei[(size_t)E + e];     // dst
  const float4* hp = (const float4*)(h + (size_t)s * DIM) + lane * 2;
  float4 a = hp[0];
  float4 b = hp[1];
  float* zp = z + (size_t)d * DIM + lane * 8;
  __hip_atomic_fetch_add(zp + 0, a.x, __ATOMIC_RELAXED, __HIP_MEMORY_SCOPE_AGENT);
  __hip_atomic_fetch_add(zp + 1, a.y, __ATOMIC_RELAXED, __HIP_MEMORY_SCOPE_AGENT);
  __hip_atomic_fetch_add(zp + 2, a.z, __ATOMIC_RELAXED, __HIP_MEMORY_SCOPE_AGENT);
  __hip_atomic_fetch_add(zp + 3, a.w, __ATOMIC_RELAXED, __HIP_MEMORY_SCOPE_AGENT);
  __hip_atomic_fetch_add(zp + 4, b.x, __ATOMIC_RELAXED, __HIP_MEMORY_SCOPE_AGENT);
  __hip_atomic_fetch_add(zp + 5, b.y, __ATOMIC_RELAXED, __HIP_MEMORY_SCOPE_AGENT);
  __hip_atomic_fetch_add(zp + 6, b.z, __ATOMIC_RELAXED, __HIP_MEMORY_SCOPE_AGENT);
  __hip_atomic_fetch_add(zp + 7, b.w, __ATOMIC_RELAXED, __HIP_MEMORY_SCOPE_AGENT);
}

// A-fragment load (optionally BN-affine + ReLU on A's columns), contiguous k order:
// lane(0-15) = row, K runs [kb,kb+16), kb = 16*(lane>=16).
template <bool BN_IN>
__device__ __forceinline__ v16bf load_afrag(const float* __restrict__ arow, int kb,
                                            const float* __restrict__ bnA,
                                            const float* __restrict__ bnC) {
  float4 f0 = *(const float4*)(arow + kb);
  float4 f1 = *(const float4*)(arow + kb + 4);
  float4 f2 = *(const float4*)(arow + kb + 8);
  float4 f3 = *(const float4*)(arow + kb + 12);
  if (BN_IN) {
    float4 a0 = *(const float4*)(bnA + kb),      c0 = *(const float4*)(bnC + kb);
    float4 a1 = *(const float4*)(bnA + kb + 4),  c1 = *(const float4*)(bnC + kb + 4);
    float4 a2 = *(const float4*)(bnA + kb + 8),  c2 = *(const float4*)(bnC + kb + 8);
    float4 a3 = *(const float4*)(bnA + kb + 12), c3 = *(const float4*)(bnC + kb + 12);
    f0.x = fmaxf(fmaf(a0.x, f0.x, c0.x), 0.f); f0.y = fmaxf(fmaf(a0.y, f0.y, c0.y), 0.f);
    f0.z = fmaxf(fmaf(a0.z, f0.z, c0.z), 0.f); f0.w = fmaxf(fmaf(a0.w, f0.w, c0.w), 0.f);
    f1.x = fmaxf(fmaf(a1.x, f1.x, c1.x), 0.f); f1.y = fmaxf(fmaf(a1.y, f1.y, c1.y), 0.f);
    f1.z = fmaxf(fmaf(a1.z, f1.z, c1.z), 0.f); f1.w = fmaxf(fmaf(a1.w, f1.w, c1.w), 0.f);
    f2.x = fmaxf(fmaf(a2.x, f2.x, c2.x), 0.f); f2.y = fmaxf(fmaf(a2.y, f2.y, c2.y), 0.f);
    f2.z = fmaxf(fmaf(a2.z, f2.z, c2.z), 0.f); f2.w = fmaxf(fmaf(a2.w, f2.w, c2.w), 0.f);
    f3.x = fmaxf(fmaf(a3.x, f3.x, c3.x), 0.f); f3.y = fmaxf(fmaf(a3.y, f3.y, c3.y), 0.f);
    f3.z = fmaxf(fmaf(a3.z, f3.z, c3.z), 0.f); f3.w = fmaxf(fmaf(a3.w, f3.w, c3.w), 0.f);
  }
  return cvt16(f0, f1, f2, f3);
}

// ---------------------------------------------------------------- GEMM
// C[N,256] = act(A) @ W + bias; per-column sum/sumsq accumulated for the next
// BatchNorm. Weights staged once per block into LDS by the Tensor Data Mover;
// one wave computes 32 rows x 128 cols (2 A-frags x 8 B-frags -> 128 WMMAs).
template <bool BN_IN>
__global__ __launch_bounds__(256) void gemm_layer(
    const float* __restrict__ A, const __bf16* __restrict__ Bt,
    const float* __restrict__ bias,
    const float* __restrict__ bnA, const float* __restrict__ bnC,
    float* __restrict__ C, float* __restrict__ gsum, float* __restrict__ gsq,
    int nRowTiles) {
  extern __shared__ char smem[];
  __bf16* Bsh  = (__bf16*)smem;                        // 256 * LDB * 2 = 135168 B
  float*  lsum = (float*)(smem + 256 * LDB * 2);       // 1 KB
  float*  lsq  = lsum + DIM;                           // 1 KB

  if (threadIdx.x < DIM) { lsum[threadIdx.x] = 0.0f; lsq[threadIdx.x] = 0.0f; }
  if (threadIdx.x < 32) {                              // wave 0 issues the DMA
    tdm_load_weights(Bt, (unsigned)(uintptr_t)Bsh);
  }
  __syncthreads();

  const int wave    = threadIdx.x >> 5;
  const int lane    = threadIdx.x & 31;
  const int half    = lane >> 4;
  const int lrow    = lane & 15;
  const int strip   = wave >> 1;          // 0..3 : which 32-row strip
  const int colHalf = wave & 1;           // 0..1 : which 128-col half
  const int rt0 = blockIdx.x * 8 + strip * 2;
  const int rt1 = rt0 + 1;
  const bool act0 = rt0 < nRowTiles;      // wave-uniform -> EXEC stays all-1
  const bool act1 = rt1 < nRowTiles;
  const int rt0c = act0 ? rt0 : nRowTiles - 1;
  const int rt1c = act1 ? rt1 : nRowTiles - 1;

  v8f acc0[8] = {};
  v8f acc1[8] = {};
  const float*  arow0 = A + (size_t)(rt0c * 16 + lrow) * DIM;
  const float*  arow1 = A + (size_t)(rt1c * 16 + lrow) * DIM;
  const __bf16* bbase = Bsh + (size_t)(colHalf * 128 + lrow) * LDB + half * 16;

#pragma unroll
  for (int kk = 0; kk < 8; ++kk) {
    const int kb = kk * 32 + half * 16;
    v16bf av0 = load_afrag<BN_IN>(arow0, kb, bnA, bnC);
    v16bf av1 = load_afrag<BN_IN>(arow1, kb, bnA, bnC);
    __builtin_prefetch(arow0 + 32 * DIM, 0, 3);        // next strip prefetch
#pragma unroll
    for (int t = 0; t < 8; ++t) {
      const v16bf bv = *(const v16bf*)(bbase + (size_t)(t * 16) * LDB + kk * 32);
      acc0[t] = __builtin_amdgcn_wmma_f32_16x16x32_bf16(
          false, av0, false, bv, (short)0, acc0[t], false, false);
      acc1[t] = __builtin_amdgcn_wmma_f32_16x16x32_bf16(
          false, av1, false, bv, (short)0, acc1[t], false, false);
    }
  }

  // Epilogue: bias, store, per-column stats (D layout: lane col = lrow,
  // VGPR i -> row i + 8*half).
  auto epilogue = [&](v8f* acc, int rtc, bool act) {
    if (!act) return;
    const int mrow0 = rtc * 16 + half * 8;
#pragma unroll
    for (int t = 0; t < 8; ++t) {
      const int col = colHalf * 128 + t * 16 + lrow;
      const float b = bias[col];
      float s = 0.0f, q = 0.0f;
#pragma unroll
      for (int i = 0; i < 8; ++i) {
        float v = acc[t][i] + b;
        C[(size_t)(mrow0 + i) * DIM + col] = v;
        s += v;
        q += v * v;
      }
      __hip_atomic_fetch_add(&lsum[col], s, __ATOMIC_RELAXED, __HIP_MEMORY_SCOPE_WORKGROUP);
      __hip_atomic_fetch_add(&lsq[col],  q, __ATOMIC_RELAXED, __HIP_MEMORY_SCOPE_WORKGROUP);
    }
  };
  epilogue(acc0, rt0c, act0);
  epilogue(acc1, rt1c, act1);

  __syncthreads();
  if (threadIdx.x < DIM) {
    __hip_atomic_fetch_add(&gsum[threadIdx.x], lsum[threadIdx.x], __ATOMIC_RELAXED, __HIP_MEMORY_SCOPE_AGENT);
    __hip_atomic_fetch_add(&gsq[threadIdx.x],  lsq[threadIdx.x],  __ATOMIC_RELAXED, __HIP_MEMORY_SCOPE_AGENT);
  }
}

// ---------------------------------------------------------------- BN helpers
__global__ void bn_coeffs(const float* __restrict__ gsum, const float* __restrict__ gsq,
                          const float* __restrict__ gamma, const float* __restrict__ beta,
                          float* __restrict__ aOut, float* __restrict__ cOut, float invN) {
  int i = threadIdx.x;
  float mu  = gsum[i] * invN;
  float var = gsq[i] * invN - mu * mu;
  float a   = gamma[i] * rsqrtf(var + BN_EPS);
  aOut[i] = a;
  cOut[i] = beta[i] - mu * a;
}

__global__ void apply_bn(const float4* __restrict__ zin, const float* __restrict__ a,
                         const float* __restrict__ c, float4* __restrict__ hdst,
                         float4* __restrict__ zdst, int doRelu, int n4) {
  int i = blockIdx.x * blockDim.x + threadIdx.x;
  if (i >= n4) return;
  int col = (i * 4) & (DIM - 1);
  float4 v  = zin[i];
  float4 av = *(const float4*)(a + col);
  float4 cv = *(const float4*)(c + col);
  float4 r;
  r.x = fmaf(av.x, v.x, cv.x);
  r.y = fmaf(av.y, v.y, cv.y);
  r.z = fmaf(av.z, v.z, cv.z);
  r.w = fmaf(av.w, v.w, cv.w);
  if (doRelu) {
    r.x = fmaxf(r.x, 0.f); r.y = fmaxf(r.y, 0.f);
    r.z = fmaxf(r.z, 0.f); r.w = fmaxf(r.w, 0.f);
  }
  hdst[i] = r;
  if (zdst) zdst[i] = r;
}

// ---------------------------------------------------------------- launcher
extern "C" void kernel_launch(void* const* d_in, const int* in_sizes, int n_in,
                              void* d_out, int out_size, void* d_ws, size_t ws_size,
                              hipStream_t stream) {
  const float*     x   = (const float*)d_in[0];
  const long long* ei  = (const long long*)d_in[1];   // int64 per reference
  const float*     W1  = (const float*)d_in[2];
  const float*     b1  = (const float*)d_in[3];
  const float*     g1  = (const float*)d_in[4];
  const float*     be1 = (const float*)d_in[5];
  const float*     W2  = (const float*)d_in[6];
  const float*     b2  = (const float*)d_in[7];
  const float*     g2  = (const float*)d_in[8];
  const float*     be2 = (const float*)d_in[9];
  float* out = (float*)d_out;

  const int N = in_sizes[0] / DIM;     // 100000
  const int E = in_sizes[1] / 2;       // 400000
  const size_t nelems = (size_t)N * DIM;
  const int n4 = (int)(nelems / 4);
  const int nRowTiles = N / 16;        // 6250 (exact)

  // Workspace layout
  float*  hbuf = (float*)d_ws;
  float*  zbuf = hbuf + nelems;
  float*  y1   = zbuf + nelems;
  __bf16* w1t  = (__bf16*)(y1 + nelems);
  __bf16* w2t  = w1t + (size_t)NLAYERS * DIM * DIM;
  float*  stats = (float*)(w2t + (size_t)NLAYERS * DIM * DIM);
  float* gsum1 = stats;            // 256
  float* gsq1  = stats + 256;
  float* gsum2 = stats + 512;
  float* gsq2  = stats + 768;
  float* a1 = stats + 1024; float* c1 = stats + 1280;
  float* a2 = stats + 1536; float* c2 = stats + 1792;

  const int wTot = NLAYERS * DIM * DIM;   // 393216
  pack_weights<<<wTot / 256, 256, 0, stream>>>(W1, w1t);
  pack_weights<<<wTot / 256, 256, 0, stream>>>(W2, w2t);

  const float invN = 1.0f / (float)N;
  const int gemmGrid = (nRowTiles + 7) / 8;           // 8 row-tiles per block
  const int scatGrid = (E * 32 + 255) / 256;
  const int elemGrid = (n4 + 255) / 256;
  const size_t gemmLds = (size_t)256 * LDB * 2 + 2 * DIM * sizeof(float);

  for (int l = 0; l < NLAYERS; ++l) {
    const float* hcur = (l == 0) ? x : hbuf;
    if (l == 0)
      copy_f4<<<elemGrid, 256, 0, stream>>>((const float4*)x, (float4*)zbuf, n4);
    zero_stats<<<1, 1024, 0, stream>>>(stats);
    // z = h + sum_{edges} h[src]  (z pre-seeded with h)
    scatter_add<<<scatGrid, 256, 0, stream>>>(hcur, ei, zbuf, E);
    // y1 = z @ W1 + b1 ; stats for inner BN
    gemm_layer<false><<<gemmGrid, 256, gemmLds, stream>>>(
        zbuf, w1t + (size_t)l * DIM * DIM, b1 + l * DIM,
        nullptr, nullptr, y1, gsum1, gsq1, nRowTiles);
    bn_coeffs<<<1, DIM, 0, stream>>>(gsum1, gsq1, g1 + l * DIM, be1 + l * DIM, a1, c1, invN);
    // z = relu(BN(y1)) @ W2 + b2 ; stats for outer BN
    gemm_layer<true><<<gemmGrid, 256, gemmLds, stream>>>(
        y1, w2t + (size_t)l * DIM * DIM, b2 + l * DIM,
        a1, c1, zbuf, gsum2, gsq2, nRowTiles);
    bn_coeffs<<<1, DIM, 0, stream>>>(gsum2, gsq2, g2 + l * DIM, be2 + l * DIM, a2, c2, invN);
    const int last = (l == NLAYERS - 1);
    apply_bn<<<elemGrid, 256, 0, stream>>>(
        (const float4*)zbuf, a2, c2,
        (float4*)(last ? out : hbuf),
        last ? nullptr : (float4*)zbuf, !last, n4);
  }
}
